// TemporalAttentionModule_25623774888591
// MI455X (gfx1250) — compile-verified, matrix operand order checked by reading, not run on hardware
//
#include <hip/hip_runtime.h>
#include <hip/hip_bf16.h>

// Problem sizes (fixed by the reference).
#define B_ 64
#define T_ 512
#define D_ 2048
#define H_ 512

typedef __attribute__((ext_vector_type(16))) __bf16 v16bf;
typedef __attribute__((ext_vector_type(8)))  float  v8f;
typedef int avec16 __attribute__((vector_size(16)));   // b128 payload type

#if defined(__AMDGCN__) && __has_builtin(__builtin_amdgcn_global_load_async_to_lds_b128) && __has_builtin(__builtin_amdgcn_s_wait_asynccnt)
#define USE_ASYNC_LDS 1
#else
#define USE_ASYNC_LDS 0
#endif

#if defined(__AMDGCN__) && __has_builtin(__builtin_amdgcn_tanhf)
__device__ __forceinline__ float fast_tanh(float x) { return __builtin_amdgcn_tanhf(x); }
#else
__device__ __forceinline__ float fast_tanh(float x) { return tanhf(x); }
#endif

__device__ __forceinline__ unsigned short f32_to_bf16(float f) {
  unsigned int u = __float_as_uint(f);
  u += 0x7FFFu + ((u >> 16) & 1u);   // round-to-nearest-even
  return (unsigned short)(u >> 16);
}

__device__ __forceinline__ unsigned pack_bf16x2(float lo, float hi) {
  return (unsigned)f32_to_bf16(lo) | ((unsigned)f32_to_bf16(hi) << 16);
}

union Frag16 { uint4 u[2]; v16bf v; };

#if USE_ASYNC_LDS
#define ASYNC_COPY16(gp, lp, off)                                                   \
  __builtin_amdgcn_global_load_async_to_lds_b128(                                   \
      (__attribute__((address_space(1))) avec16*)(unsigned long long)(gp),          \
      (__attribute__((address_space(3))) avec16*)(unsigned)(unsigned long long)(lp),\
      (off), 0)
#endif

// Stage 32 consecutive floats (one thread's slice of a 32x256 chunk) from
// global into the fp32 LDS staging buffer. Async (ASYNCcnt) when available.
__device__ __forceinline__ void stage_chunk(const float* __restrict__ g,
                                            float* __restrict__ l) {
#if USE_ASYNC_LDS
  ASYNC_COPY16(g, l, 0);
  ASYNC_COPY16(g, l, 16);
  ASYNC_COPY16(g, l, 32);
  ASYNC_COPY16(g, l, 48);
  ASYNC_COPY16(g, l, 64);
  ASYNC_COPY16(g, l, 80);
  ASYNC_COPY16(g, l, 96);
  ASYNC_COPY16(g, l, 112);
#else
  #pragma unroll
  for (int e = 0; e < 8; ++e)
    *reinterpret_cast<float4*>(l + e * 4) =
        *reinterpret_cast<const float4*>(g + e * 4);
#endif
}

// ---------------------------------------------------------------------------
// Kernel 1: Uh[b,h] = hidden_text[b,:] @ Ua[:,h] + ba[h]   (tiny, fp32 VALU)
// ---------------------------------------------------------------------------
__global__ __launch_bounds__(256) void uh_kernel(
    const float* __restrict__ ht, const float* __restrict__ Ua,
    const float* __restrict__ ba, float* __restrict__ uh) {
  const int h = blockIdx.x * 256 + threadIdx.x;
  const int b = blockIdx.y;
  const float* hrow = ht + (size_t)b * D_;
  float acc = 0.f;
  for (int d = 0; d < D_; ++d)
    acc = fmaf(hrow[d], Ua[(size_t)d * H_ + h], acc);
  uh[b * H_ + h] = acc + ba[h];
}

// ---------------------------------------------------------------------------
// Kernel 2: WaT[h][k] = bf16(Wa[k][h])  — 2 MB, stays L2-resident for the GEMM
// ---------------------------------------------------------------------------
__global__ __launch_bounds__(256) void wa_transpose_bf16_kernel(
    const float* __restrict__ Wa, unsigned short* __restrict__ waT) {
  const int idx = blockIdx.x * 256 + threadIdx.x;  // 0 .. D_*H_-1
  const int d = idx >> 9;      // / H_
  const int h = idx & (H_ - 1);
  waT[(size_t)h * D_ + d] = f32_to_bf16(Wa[idx]);
}

// ---------------------------------------------------------------------------
// Kernel 3: fused  scores[m] = Va . tanh(hidden_frames[m,:] @ Wa + Uh[b,:])
//   Workgroup = 256 threads (8 waves), owns 32 M-rows x full H=512.
//   A chunk (32x256 fp32) double-buffered into LDS via ASYNCcnt async loads,
//   converted to bf16 in LDS, consumed by v_wmma_f32_16x16x32_bf16.
//   B (WaT bf16, L2-resident) streamed with WMMA per-lane layout addressing.
// ---------------------------------------------------------------------------
__global__ __launch_bounds__(256) void scores_kernel(
    const float* __restrict__ hf, const unsigned short* __restrict__ waT,
    const float* __restrict__ uh, const float* __restrict__ va,
    float* __restrict__ scores) {
  __shared__ float f32Buf[2][32 * 256];      // 64 KB async staging
  __shared__ unsigned short aLds[32 * 264];  // bf16 A tile, +8 pad/row
  __shared__ float uhLds[H_];
  __shared__ float vaLds[H_];
  __shared__ float scoreLds[32];

  const int tid   = threadIdx.x;
  const int lane  = tid & 31;
  const int wave  = tid >> 5;
  const int mBase = blockIdx.x * 32;
  const int b     = mBase >> 9;        // / T_

  if (tid < 32) scoreLds[tid] = 0.f;
  for (int i = tid; i < H_; i += 256) {
    uhLds[i] = uh[b * H_ + i];
    vaLds[i] = va[i];
  }

  const int col   = lane & 15;   // matrix row (A) / col (B) within 16-subtile
  const int half  = lane >> 4;   // selects the +8 K-run per ISA layout
  const int nBase = wave * 64;   // this wave's 64 H-columns (4 x 16)

  // Staging slice: thread -> row rS = tid/8, cols [cS, cS+32)
  const int rS = tid >> 3;
  const int cS = (tid & 7) * 32;
  const float* gRow = hf + (size_t)(mBase + rS) * D_ + cS;
  const int lOff = rS * 256 + cS;

  v8f acc[2][4];
  {
    v8f zero = {};
    #pragma unroll
    for (int i = 0; i < 2; ++i)
      #pragma unroll
      for (int j = 0; j < 4; ++j) acc[i][j] = zero;
  }

  const unsigned short* bBase = waT + (size_t)(nBase + col) * D_ + half * 8;

  stage_chunk(gRow, &f32Buf[0][lOff]);            // prologue: chunk 0

  for (int kc = 0; kc < 8; ++kc) {                // K chunks of 256
    const int k0 = kc * 256;
    if (kc < 7) {
      stage_chunk(gRow + (kc + 1) * 256, &f32Buf[(kc + 1) & 1][lOff]);
#if USE_ASYNC_LDS
      __builtin_amdgcn_s_wait_asynccnt(8);        // chunk kc resident
#endif
    } else {
#if USE_ASYNC_LDS
      __builtin_amdgcn_s_wait_asynccnt(0);
#endif
    }
    __syncthreads();

    // Convert fp32 staging chunk -> bf16 aLds (coalesced, conflict-free).
    #pragma unroll
    for (int e = 0; e < 8; ++e) {
      const int idx = e * 1024 + tid * 4;
      const int r = idx >> 8;
      const int c = idx & 255;
      const float4 f = *reinterpret_cast<const float4*>(&f32Buf[kc & 1][idx]);
      uint2 p;
      p.x = pack_bf16x2(f.x, f.y);
      p.y = pack_bf16x2(f.z, f.w);
      *reinterpret_cast<uint2*>(&aLds[r * 264 + c]) = p;
    }
    __syncthreads();

    #pragma unroll
    for (int ks = 0; ks < 8; ++ks) {              // 8 WMMA k-steps of 32
      const int kk = ks * 32;
      Frag16 aF[2];
      #pragma unroll
      for (int i = 0; i < 2; ++i) {
        // 16-bit A layout: lane half 0 -> K {0..7,16..23}, half 1 -> +8
        const unsigned short* p = &aLds[(i * 16 + col) * 264 + kk + half * 8];
        aF[i].u[0] = *reinterpret_cast<const uint4*>(p);
        aF[i].u[1] = *reinterpret_cast<const uint4*>(p + 16);
      }
      #pragma unroll
      for (int j = 0; j < 4; ++j) {
        Frag16 bF;
        const unsigned short* p = bBase + (size_t)j * 16 * D_ + k0 + kk;
        bF.u[0] = *reinterpret_cast<const uint4*>(p);
        bF.u[1] = *reinterpret_cast<const uint4*>(p + 16);
        acc[0][j] = __builtin_amdgcn_wmma_f32_16x16x32_bf16(
            false, aF[0].v, false, bF.v, (short)0, acc[0][j], false, false);
        acc[1][j] = __builtin_amdgcn_wmma_f32_16x16x32_bf16(
            false, aF[1].v, false, bF.v, (short)0, acc[1][j], false, false);
      }
    }
  }

  // Epilogue: scores[m] += sum_h Va[h] * tanh(ws[m,h] + Uh[b,h])
  // C/D layout: VGPR v, lanes 0-15 -> M=v ; lanes 16-31 -> M=8+v ; N = lane%16
  #pragma unroll
  for (int i = 0; i < 2; ++i) {
    #pragma unroll
    for (int v = 0; v < 8; ++v) {
      float p = 0.f;
      #pragma unroll
      for (int j = 0; j < 4; ++j) {
        const int h = nBase + j * 16 + col;
        p += vaLds[h] * fast_tanh(acc[i][j][v] + uhLds[h]);
      }
      // reduce across the 16 lanes of this half (wave32 shuffles)
      p += __shfl_xor(p, 1);
      p += __shfl_xor(p, 2);
      p += __shfl_xor(p, 4);
      p += __shfl_xor(p, 8);
      if (col == 0)
        atomicAdd(&scoreLds[i * 16 + half * 8 + v], p);
    }
  }
  __syncthreads();
  if (tid < 32) scores[mBase + tid] = scoreLds[tid];
}

// ---------------------------------------------------------------------------
// Kernel 4: softmax over T per batch (one block per b)
// ---------------------------------------------------------------------------
__global__ __launch_bounds__(512) void softmax_kernel(
    const float* __restrict__ scores, float* __restrict__ att) {
  __shared__ float red[T_];
  const int b = blockIdx.x, tid = threadIdx.x;
  const float s = scores[b * T_ + tid];
  red[tid] = s;
  __syncthreads();
  for (int off = 256; off > 0; off >>= 1) {
    if (tid < off) red[tid] = fmaxf(red[tid], red[tid + off]);
    __syncthreads();
  }
  const float mx = red[0];
  __syncthreads();
  const float e = __expf(s - mx);
  red[tid] = e;
  __syncthreads();
  for (int off = 256; off > 0; off >>= 1) {
    if (tid < off) red[tid] += red[tid + off];
    __syncthreads();
  }
  att[b * T_ + tid] = e / red[0];
}

// ---------------------------------------------------------------------------
// Kernel 5: out[b,d] = sum_t att[b,t] * hidden_frames[b,t,d]  (2nd HBM pass)
// ---------------------------------------------------------------------------
__global__ __launch_bounds__(256) void weighted_sum_kernel(
    const float* __restrict__ att, const float* __restrict__ hf,
    float* __restrict__ out) {
  const int d = blockIdx.x * 256 + threadIdx.x;
  const int b = blockIdx.y;
  const float* base = hf + (size_t)b * T_ * D_ + d;
  const float* arow = att + b * T_;
  float a0 = 0.f, a1 = 0.f, a2 = 0.f, a3 = 0.f;
  for (int t = 0; t < T_; t += 4) {
    a0 = fmaf(arow[t + 0], base[(size_t)(t + 0) * D_], a0);
    a1 = fmaf(arow[t + 1], base[(size_t)(t + 1) * D_], a1);
    a2 = fmaf(arow[t + 2], base[(size_t)(t + 2) * D_], a2);
    a3 = fmaf(arow[t + 3], base[(size_t)(t + 3) * D_], a3);
  }
  out[(size_t)b * D_ + d] = (a0 + a1) + (a2 + a3);
}

// ---------------------------------------------------------------------------
extern "C" void kernel_launch(void* const* d_in, const int* in_sizes, int n_in,
                              void* d_out, int out_size, void* d_ws, size_t ws_size,
                              hipStream_t stream) {
  const float* hf = (const float*)d_in[0];  // hidden_frames [B,T,D]
  const float* ht = (const float*)d_in[1];  // hidden_text   [B,D]
  const float* Wa = (const float*)d_in[2];  // [D,H]
  const float* Ua = (const float*)d_in[3];  // [D,H]
  const float* Va = (const float*)d_in[4];  // [H]
  const float* ba = (const float*)d_in[5];  // [1,1,H]
  float* out = (float*)d_out;               // [B,D]

  char* ws = (char*)d_ws;
  unsigned short* waT = (unsigned short*)ws;                        // 2 MB
  float* uh     = (float*)(ws + 2u * 1024 * 1024);                  // 128 KB
  float* scores = (float*)(ws + 2u * 1024 * 1024 + 128u * 1024);    // 128 KB
  float* att    = (float*)(ws + 2u * 1024 * 1024 + 256u * 1024);    // 128 KB

  uh_kernel<<<dim3(H_ / 256, B_), 256, 0, stream>>>(ht, Ua, ba, uh);
  wa_transpose_bf16_kernel<<<(D_ * H_) / 256, 256, 0, stream>>>(Wa, waT);
  scores_kernel<<<(B_ * T_) / 32, 256, 0, stream>>>(hf, waT, uh, Va, scores);
  softmax_kernel<<<B_, T_, 0, stream>>>(scores, att);
  weighted_sum_kernel<<<dim3(D_ / 256, B_), 256, 0, stream>>>(att, hf, out);
}
